// NeuralAlgorithm_18597208392400
// MI455X (gfx1250) — matrix operand order repaired and checked by reading, not compiled, not run on hardware
//
#include <hip/hip_runtime.h>

typedef __attribute__((ext_vector_type(16))) _Float16 v16h;
typedef __attribute__((ext_vector_type(8)))  _Float16 v8h;
typedef __attribute__((ext_vector_type(8)))  float    v8f;

#define B_   32
#define H_   256
#define W_   256
#define C_   16
#define HID_ 128
#define K9C  144          // 9*C
#define WAVES 4
#define TILES 2           // 16-pixel tiles per wave
#define PIX_PER_BLOCK (WAVES * TILES * 16)   // 128
#define NBW  (PIX_PER_BLOCK + 2)             // 130 (halo)

#define SHUF16(lo, hi) __builtin_shufflevector(lo, hi, 0,1,2,3,4,5,6,7,8,9,10,11,12,13,14,15)

__global__ __launch_bounds__(128)
void nca_step_kernel(const float* __restrict__ x,
                     const float* __restrict__ W1,
                     const float* __restrict__ b1,
                     const float* __restrict__ W2,
                     const float* __restrict__ b2,
                     float* __restrict__ out)
{
    // Static LDS: 32768 + 4096 + 32 + 4096 + 16384(UN) + 1584 + 512 = 59,472 B
    __shared__ __align__(16) _Float16 W1c[8 * 4 * 32 * 16];   // B1 frags kc=0..3 (swizzled)
    __shared__ __align__(16) _Float16 W1t[8 * 16 * 16];       // B1 tail kc=4, [nt][n][16 k], lane half picks 8
    __shared__ __align__(16) _Float16 Zblk[16];               // zeros for tail elements 8..15, all lanes
    __shared__ __align__(16) _Float16 W2s[4 * 32 * 16];       // B2 frags (swizzled)
    __shared__ __align__(16) unsigned char UN[WAVES * 16 * HID_ * 2]; // NB phase -> Hs phase
    __shared__ float AL[3][132];                              // alive plane
    __shared__ float CLR[WAVES][TILES * 16];                  // cleared flags

    _Float16* const NB = (_Float16*)UN;   // [3][NBW*C_] = 12,480 B (dies at 2nd barrier)
    _Float16* const Hs = (_Float16*)UN;   // [WAVES][16*HID_] = 16,384 B (after 2nd barrier)

    const int tid  = threadIdx.x;
    const int wave = tid >> 5;
    const int lane = tid & 31;
    const int m    = lane & 15;   // A/D "M" row / column within N-tile
    const int hs   = lane >> 4;   // lane-half selector
    const int cb   = hs * 8;      // channel/K base for this lane-half

    const int x0 = blockIdx.x * PIX_PER_BLOCK;
    const int y  = blockIdx.y;
    const int bb = blockIdx.z;

    // ---- stage W1 (coalesced global reads, LDS scatter to B-fragment order) ----
    for (int s = tid; s < K9C * HID_; s += 128) {
        int K = s >> 7;                   // 0..143
        int n = s & 127;
        _Float16 v = (_Float16)W1[s];
        int nt  = n >> 4, c16 = n & 15;
        int kc  = K >> 5, rem = K & 31;
        int hsel = rem >> 4, i = rem & 15;
        if (kc < 4)
            W1c[(((nt << 2) | kc) * 32 + (((rem >> 3) & 1) << 4) + c16) * 16 + ((rem & 7) | (hsel << 3))] = v;
        else                               // K=128..143 -> rem<16, hsel==0
            W1t[((nt << 4) + c16) * 16 + i] = v;
    }
    if (tid < 16) Zblk[tid] = (_Float16)0.f;

    // ---- stage W2 to B-fragment order ----
    for (int s = tid; s < HID_ * C_; s += 128) {
        int K = s >> 4, n = s & 15;
        int kc = K >> 5, rem = K & 31;
        int hsel = rem >> 4, i = rem & 15;
        W2s[(kc * 32 + (((rem >> 3) & 1) << 4) + n) * 16 + ((rem & 7) | (hsel << 3))] = (_Float16)W2[s];
    }

    // ---- stage 3 rows x 130 px x 16 ch neighborhood + alive plane ----
    for (int i = tid; i < 3 * NBW * C_; i += 128) {
        int r   = i / (NBW * C_);
        int rem = i - r * (NBW * C_);
        int px  = rem >> 4;
        int c   = rem & 15;
        int gy  = y + r - 1;
        int gx  = x0 + px - 1;
        float v = 0.f;
        if (gy >= 0 && gy < H_ && gx >= 0 && gx < W_)
            v = x[(((bb * H_ + gy) * W_) + gx) * C_ + c];
        NB[r * (NBW * C_) + rem] = (_Float16)v;
        if (c == 3) AL[r][px] = (v > 0.1f) ? 1.f : 0.f;   // threshold on fp32
    }
    __syncthreads();

    // ---- cleared flags: each lane handles one of the wave's 32 pixels ----
    {
        int pp = wave * 32 + lane;
        float ns = 0.f;
        #pragma unroll
        for (int r = 0; r < 3; ++r)
            #pragma unroll
            for (int j = 0; j < 3; ++j)
                ns += AL[r][pp + j];
        CLR[wave][lane] = (ns == 0.f) ? 1.f : 0.f;
    }

    // ---- A1 fragments for both tiles (NB dies after this) ----
    static const int RJ[9][2] = {{0,0},{0,1},{0,2},{1,0},{1,1},{1,2},{2,0},{2,1},{2,2}};
    v16h a1[TILES][5];
    #pragma unroll
    for (int t = 0; t < TILES; ++t) {
        int p = wave * 32 + t * 16 + m;   // pixel (A row M) for this lane
        #pragma unroll
        for (int kc = 0; kc < 4; ++kc) {
            int n0 = 2 * kc, n1 = 2 * kc + 1;
            v8h lo = *(const v8h*)&NB[(RJ[n0][0] * NBW + p + RJ[n0][1]) * C_ + cb];
            v8h hi = *(const v8h*)&NB[(RJ[n1][0] * NBW + p + RJ[n1][1]) * C_ + cb];
            a1[t][kc] = SHUF16(lo, hi);
        }
        v8h lo = *(const v8h*)&NB[(2 * NBW + p + 2) * C_ + cb];   // neighbor 8
        v8h z  = {};
        a1[t][4] = SHUF16(lo, z);                                  // K>=144 zeros
    }
    __syncthreads();     // UN becomes Hs

    // tail fragment pointers (uniform across nt except the nt<<4 term)
    const float bias2 = b2[m];             // m == output channel in epilogue

    // ---- per-tile: GEMM1 -> Hs -> GEMM2 -> epilogue ----
    #pragma unroll
    for (int t = 0; t < TILES; ++t) {
        // GEMM1: [16x144] @ [144x128]; process nt-tiles in pairs with all B
        // fragments preloaded -> deep DS pipeline + two independent WMMA chains.
        #pragma unroll
        for (int nt2 = 0; nt2 < 4; ++nt2) {
            const int ntA = nt2 * 2, ntB = nt2 * 2 + 1;
            v16h bfA[5], bfB[5];
            #pragma unroll
            for (int kc = 0; kc < 4; ++kc) {
                const _Float16* pA = &W1c[(((ntA << 2) | kc) * 32 + lane) * 16];
                const _Float16* pB = &W1c[(((ntB << 2) | kc) * 32 + lane) * 16];
                bfA[kc] = SHUF16(*(const v8h*)pA, *(const v8h*)(pA + 8));
                bfB[kc] = SHUF16(*(const v8h*)pB, *(const v8h*)(pB + 8));
            }
            {   // kc==4 tail: elems 0..7 from W1t at +cb, elems 8..15 -> zero block
                const _Float16* lA = &W1t[((ntA << 4) + m) * 16 + cb];
                const _Float16* hA = &Zblk[8];
                const _Float16* lB = &W1t[((ntB << 4) + m) * 16 + cb];
                const _Float16* hB = &Zblk[8];
                bfA[4] = SHUF16(*(const v8h*)lA, *(const v8h*)hA);
                bfB[4] = SHUF16(*(const v8h*)lB, *(const v8h*)hB);
            }
            v8f accA = {}, accB = {};
            #pragma unroll
            for (int kc = 0; kc < 5; ++kc) {
                accA = __builtin_amdgcn_wmma_f32_16x16x32_f16(
                           false, a1[t][kc], false, bfA[kc], (short)0, accA, false, false);
                accB = __builtin_amdgcn_wmma_f32_16x16x32_f16(
                           false, a1[t][kc], false, bfB[kc], (short)0, accB, false, false);
            }
            float biasA = b1[ntA * 16 + m];
            float biasB = b1[ntB * 16 + m];
            #pragma unroll
            for (int v = 0; v < 8; ++v) {                  // D-layout scatter stores
                float hA = accA[v] + biasA; hA = hA > 0.f ? hA : 0.f;
                float hB = accB[v] + biasB; hB = hB > 0.f ? hB : 0.f;
                Hs[(wave * 16 + hs * 8 + v) * HID_ + ntA * 16 + m] = (_Float16)hA;
                Hs[(wave * 16 + hs * 8 + v) * HID_ + ntB * 16 + m] = (_Float16)hB;
            }
        }

        // GEMM2: [16x128] @ [128x16]; preload all fragments, then 4 WMMAs
        v16h a2[4], bf2[4];
        #pragma unroll
        for (int kc = 0; kc < 4; ++kc) {
            const _Float16* ap = &Hs[(wave * 16 + m) * HID_ + kc * 32 + cb];
            a2[kc] = SHUF16(*(const v8h*)ap, *(const v8h*)(ap + 16));
            const _Float16* bp = &W2s[(kc * 32 + lane) * 16];
            bf2[kc] = SHUF16(*(const v8h*)bp, *(const v8h*)(bp + 8));
        }
        v8f acc2 = {};
        #pragma unroll
        for (int kc = 0; kc < 4; ++kc)
            acc2 = __builtin_amdgcn_wmma_f32_16x16x32_f16(
                       false, a2[kc], false, bf2[kc], (short)0, acc2, false, false);

        // epilogue: mask, bias, residual, clip, store
        #pragma unroll
        for (int v = 0; v < 8; ++v) {
            int mo = hs * 8 + v;                            // row within tile
            int gx = x0 + wave * 32 + t * 16 + mo;
            float upd = acc2[v] + bias2;
            if (CLR[wave][t * 16 + mo] != 0.f) upd = 0.f;
            long idx = ((((long)bb * H_ + y) * W_) + gx) * C_ + m;
            float o = x[idx] + upd;
            o = o < -1.f ? -1.f : (o > 1.f ? 1.f : o);
            out[idx] = o;
        }
    }
}

extern "C" void kernel_launch(void* const* d_in, const int* in_sizes, int n_in,
                              void* d_out, int out_size, void* d_ws, size_t ws_size,
                              hipStream_t stream) {
    const float* x  = (const float*)d_in[0];
    const float* W1 = (const float*)d_in[1];
    const float* b1 = (const float*)d_in[2];
    const float* W2 = (const float*)d_in[3];
    const float* b2 = (const float*)d_in[4];
    float* out = (float*)d_out;

    dim3 grid(W_ / PIX_PER_BLOCK, H_, B_);   // (2, 256, 32)
    nca_step_kernel<<<grid, 128, 0, stream>>>(x, W1, b1, W2, b2, out);
}
